// ExposureManager_46299747451382
// MI455X (gfx1250) — matrix operand order repaired and checked by reading, not compiled, not run on hardware
//
#include <hip/hip_runtime.h>
#include <hip/hip_bf16.h>
#include <math.h>

#define NG      18
#define NITEMS  20000000
#define EPSF    1e-8f
#define NBINS   (1u << 21)          // positive float bits >> 10 => index < 2^21
#define CHUNK   2048
#define NCHUNK  (NBINS / CHUNK)     // 1024

typedef __attribute__((ext_vector_type(16))) _Float16 v16h;
typedef __attribute__((ext_vector_type(8)))  float    v8f;

// Async load-to-LDS (gfx1250): guarded so absence of the builtin still compiles.
#if defined(__has_builtin)
#  if __has_builtin(__builtin_amdgcn_global_load_async_to_lds_b32) && \
      __has_builtin(__builtin_amdgcn_s_wait_asynccnt)
#    define USE_ASYNC_LDS 1
#  endif
#endif
#ifndef USE_ASYNC_LDS
#  define USE_ASYNC_LDS 0
#endif

// ---------------------------------------------------------------- zero ws
__global__ void k_zero(unsigned* __restrict__ cnt, float* __restrict__ acc) {
    unsigned stride = gridDim.x * blockDim.x;
    for (unsigned i = blockIdx.x * blockDim.x + threadIdx.x; i < NBINS; i += stride)
        cnt[i] = 0u;
    if (blockIdx.x == 0 && threadIdx.x == 0) {
        acc[0] = 0.f;                 // exact total sum of (x+eps)
        ((unsigned*)acc)[1] = 0u;     // coverage count
    }
}

// ---------------------------------------------------------------- histogram pass: ONE u32 atomic per item
__global__ void k_hist(const float* __restrict__ items, unsigned* __restrict__ cnt,
                       float* __restrict__ acc) {
    __shared__ float    sSum;
    __shared__ unsigned sCov;
    if (threadIdx.x == 0) { sSum = 0.f; sCov = 0u; }
    __syncthreads();

    const float4* it4 = (const float4*)items;
    const int n4 = NITEMS / 4;
    const int stride = gridDim.x * blockDim.x;
    float    lSum = 0.f;
    unsigned lCov = 0u;

    for (int i = blockIdx.x * blockDim.x + threadIdx.x; i < n4; i += stride) {
        __builtin_prefetch(it4 + i + stride, 0, 0);  // global_prefetch_b8
        float4 v = it4[i];
        float xs0 = v.x, xs1 = v.y, xs2 = v.z, xs3 = v.w;
        #pragma unroll
        for (int j = 0; j < 4; j++) {
            float x = (j == 0) ? xs0 : (j == 1) ? xs1 : (j == 2) ? xs2 : xs3;
            float y = x + EPSF;                       // strictly positive
            unsigned b = __float_as_uint(y) >> 10;    // monotone for positive floats
            atomicAdd(&cnt[b], 1u);                   // L2-resident, single RMW/item
            lSum += y;
            lCov += (x > 0.f) ? 1u : 0u;
        }
    }
    atomicAdd(&sSum, lSum);
    atomicAdd(&sCov, lCov);
    __syncthreads();
    if (threadIdx.x == 0) {
        atomicAdd(&acc[0], sSum);
        atomicAdd(((unsigned*)acc) + 1, sCov);
    }
}

// ---------------------------------------------------------------- per-chunk scan of bin counts
// S_b := n_b * midpoint(bin b);  chunkT = sum_b [ exclPrefix_b*S_b + S_b*(n_b+1)/2 ]
__global__ void k_chunkscan(const unsigned* __restrict__ cnt,
                            double* __restrict__ chT, double* __restrict__ chS,
                            unsigned* __restrict__ chC) {
    __shared__ unsigned sb[256];
    __shared__ double   red[256];
#if USE_ASYNC_LDS
    __shared__ unsigned tileBuf[2][256];
#endif
    const int tid  = threadIdx.x;
    const int base = blockIdx.x * CHUNK;
    const int NT   = CHUNK / 256;     // 8 tiles

    unsigned running = 0u;
    double myT = 0.0, myS = 0.0;

#if USE_ASYNC_LDS
    // kick tile 0
    __builtin_amdgcn_global_load_async_to_lds_b32(
        (int*)(cnt + base + tid), (int*)&tileBuf[0][tid], 0, 0);
#endif

    for (int t = 0; t < NT; ++t) {
        int b = base + t * 256 + tid;
#if USE_ASYNC_LDS
        if (t + 1 < NT) {
            __builtin_amdgcn_global_load_async_to_lds_b32(
                (int*)(cnt + b + 256), (int*)&tileBuf[(t + 1) & 1][tid], 0, 0);
            __builtin_amdgcn_s_wait_asynccnt(1);   // tile t landed (in-order)
        } else {
            __builtin_amdgcn_s_wait_asynccnt(0);
        }
        unsigned nb = tileBuf[t & 1][tid];          // each lane reads its own element
#else
        unsigned nb = cnt[b];
#endif
        // representative value = midpoint of the bin's value interval
        float lo  = __uint_as_float((unsigned)b << 10);
        float hi  = __uint_as_float(((unsigned)b + 1u) << 10);
        double Sb = (nb > 0u) ? (double)nb * 0.5 * ((double)lo + (double)hi) : 0.0;

        sb[tid] = nb;
        __syncthreads();
        for (int off = 1; off < 256; off <<= 1) {   // Hillis-Steele inclusive scan
            unsigned v = (tid >= off) ? sb[tid - off] : 0u;
            __syncthreads();
            sb[tid] += v;
            __syncthreads();
        }
        unsigned incl = sb[tid];
        unsigned tot  = sb[255];
        unsigned excl = incl - nb;

        myT += (double)(running + excl) * Sb + Sb * ((double)nb + 1.0) * 0.5;
        myS += Sb;
        running += tot;
        __syncthreads();
    }

    red[tid] = myT; __syncthreads();
    for (int off = 128; off >= 1; off >>= 1) { if (tid < off) red[tid] += red[tid + off]; __syncthreads(); }
    if (tid == 0) chT[blockIdx.x] = red[0];
    __syncthreads();
    red[tid] = myS; __syncthreads();
    for (int off = 128; off >= 1; off >>= 1) { if (tid < off) red[tid] += red[tid + off]; __syncthreads(); }
    if (tid == 0) { chS[blockIdx.x] = red[0]; chC[blockIdx.x] = running; }
}

// ---------------------------------------------------------------- WMMA A-fragment K index (16-bit A, 16x32, wave32)
__device__ __forceinline__ int a_kidx(int lane, int e) {
    int half = lane >> 4;
    int j = e >> 1, p = e & 1;
    return (j < 4) ? (2 * j + 8 * half + p) : (16 + 2 * (j - 4) + 8 * half + p);
}

// ---------------------------------------------------------------- finalize: 1 wave (32 lanes), WMMA MLP
__global__ void k_final(const float* __restrict__ gcounts,
                        const double* __restrict__ chT, const double* __restrict__ chS,
                        const unsigned* __restrict__ chC, const float* __restrict__ acc,
                        const float* __restrict__ W1f, const float* __restrict__ b1f,
                        const float* __restrict__ lng, const float* __restrict__ lnb,
                        const float* __restrict__ W2f, const float* __restrict__ b2f,
                        const float* __restrict__ W3f, const float* __restrict__ b3f,
                        const float* __restrict__ Wa1, const float* __restrict__ ba1,
                        const float* __restrict__ Wa2, const float* __restrict__ ba2,
                        const float* __restrict__ Wa3, const float* __restrict__ ba3,
                        float* __restrict__ out) {
    __shared__ float W1p[64 * 32];   // W1 zero-padded to K=32
    __shared__ float W3p[32 * 32];   // W3 zero-padded to M=32
    __shared__ float stateLds[32];
    __shared__ float hbuf[64];
    __shared__ float h2buf[32];
    __shared__ float mainAdj[32];
    __shared__ float normg[NG];
    __shared__ float itemGiniSh;

    const int lane = threadIdx.x;
    const int m    = lane & 15;
    const int half = lane >> 4;
    const float bmaskF = (m == 0) ? 1.f : 0.f;   // column-0 broadcast mask (no branch)

    // ---- cooperative staging of padded weights into LDS (uniform, branch-light)
    for (int idx = lane; idx < 64 * 32; idx += 32) {
        int row = idx >> 5, col = idx & 31;
        W1p[idx] = (col < 21) ? W1f[row * 21 + col] : 0.f;
    }
    for (int idx = lane; idx < 32 * 32; idx += 32) {
        int row = idx >> 5, col = idx & 31;
        W3p[idx] = (row < NG) ? W3f[row * 32 + col] : 0.f;
    }

    // ---- item gini via cross-chunk prefix + genre state (lane 0, trivial serial work)
    if (lane == 0) {
        double T = 0.0, baseCnt = 0.0;
        for (int c = 0; c < NCHUNK; ++c) {
            T += chT[c] + baseCnt * chS[c];
            baseCnt += (double)chC[c];
        }
        const double n    = (double)NITEMS;
        const double sumY = (double)acc[0];
        double g = 2.0 * T / (n * sumY) - (n + 1.0) / n;
        g = fmin(fmax(g, 0.0), 1.0);
        itemGiniSh = (float)g;

        float cov = (float)(((const unsigned*)acc)[1]) / (float)NITEMS;

        float tot = 0.f;
        for (int i = 0; i < NG; ++i) tot += gcounts[i];
        tot += EPSF;
        float divers = 0.f;
        for (int i = 0; i < NG; ++i) {
            normg[i] = gcounts[i] / tot;
            float p = normg[i] + EPSF;
            divers -= p * logf(p);
        }
        // exact 18-element gini via average-rank (identical sum to the sort formula)
        float ge[NG], sY = 0.f;
        for (int i = 0; i < NG; ++i) { ge[i] = gcounts[i] + EPSF; sY += ge[i]; }
        float Tg = 0.f;
        for (int j = 0; j < NG; ++j) {
            int less = 0, eq = 0;
            for (int k = 0; k < NG; ++k) { less += (ge[k] < ge[j]); eq += (ge[k] == ge[j]); }
            Tg += ge[j] * ((float)less + ((float)eq + 1.f) * 0.5f);
        }
        float gg = 2.f * Tg / ((float)NG * sY) - ((float)NG + 1.f) / (float)NG;
        gg = fminf(fmaxf(gg, 0.f), 1.f);

        for (int i = 0; i < NG; ++i) stateLds[i] = normg[i];
        stateLds[18] = gg;
        stateLds[19] = cov;
        stateLds[20] = divers;
        for (int i = 21; i < 32; ++i) stateLds[i] = 0.f;
    }
    __syncthreads();

    // ---- layer 1: h[64] = relu(W1[64,21] @ state + b1)   (4 WMMA tiles, K padded to 32)
    for (int t = 0; t < 4; ++t) {
        v16h a, b;
        #pragma unroll
        for (int e = 0; e < 16; ++e) {
            int k  = a_kidx(lane, e);
            a[e] = (_Float16)W1p[(t * 16 + m) * 32 + k];
            int kb = a_kidx(lane, e);
            b[e] = (_Float16)(bmaskF * stateLds[kb]);        // state in column N=0
        }
        v8f c = {};
        c = __builtin_amdgcn_wmma_f32_16x16x32_f16(false, a, false, b, (short)0, c, false, false);
        if (m == 0) {                     // column 0: lane0 -> M=0..7, lane16 -> M=8..15
            int rbase = t * 16 + half * 8;
            #pragma unroll
            for (int j = 0; j < 8; ++j)
                hbuf[rbase + j] = fmaxf(c[j] + b1f[rbase + j], 0.f);
        }
    }
    __syncthreads();

    // ---- layer norm over h[64] (lane 0, trivial)
    if (lane == 0) {
        float mu = 0.f;
        for (int i = 0; i < 64; ++i) mu += hbuf[i];
        mu *= (1.f / 64.f);
        float var = 0.f;
        for (int i = 0; i < 64; ++i) { float d = hbuf[i] - mu; var += d * d; }
        var *= (1.f / 64.f);
        float inv = 1.f / sqrtf(var + 1e-5f);
        for (int i = 0; i < 64; ++i)
            hbuf[i] = (hbuf[i] - mu) * inv * lng[i] + lnb[i];
    }
    __syncthreads();

    // ---- layer 2: h2[32] = relu(W2[32,64] @ h + b2)   (2 M-tiles x 2 K-chunks, accumulating)
    for (int t = 0; t < 2; ++t) {
        v8f c = {};
        for (int ks = 0; ks < 2; ++ks) {
            v16h a, b;
            #pragma unroll
            for (int e = 0; e < 16; ++e) {
                int k  = a_kidx(lane, e);
                a[e] = (_Float16)W2f[(t * 16 + m) * 64 + ks * 32 + k];
                int kb = a_kidx(lane, e);
                b[e] = (_Float16)(bmaskF * hbuf[ks * 32 + kb]);
            }
            c = __builtin_amdgcn_wmma_f32_16x16x32_f16(false, a, false, b, (short)0, c, false, false);
        }
        if (m == 0) {
            int rbase = t * 16 + half * 8;
            #pragma unroll
            for (int j = 0; j < 8; ++j)
                h2buf[rbase + j] = fmaxf(c[j] + b2f[rbase + j], 0.f);
        }
    }
    __syncthreads();

    // ---- layer 3: main_adj[18] = sigmoid(W3[18,32] @ h2 + b3)   (2 M-tiles, padded rows)
    for (int t = 0; t < 2; ++t) {
        v16h a, b;
        #pragma unroll
        for (int e = 0; e < 16; ++e) {
            int k = a_kidx(lane, e);
            a[e] = (_Float16)W3p[(t * 16 + m) * 32 + k];
            int kb = a_kidx(lane, e);
            b[e] = (_Float16)(bmaskF * h2buf[kb]);
        }
        v8f c = {};
        c = __builtin_amdgcn_wmma_f32_16x16x32_f16(false, a, false, b, (short)0, c, false, false);
        if (m == 0) {
            int rbase = t * 16 + half * 8;
            #pragma unroll
            for (int j = 0; j < 8; ++j) {
                int row = rbase + j;
                if (row < NG)
                    mainAdj[row] = 1.f / (1.f + expf(-(c[j] + b3f[row])));
            }
        }
    }
    __syncthreads();

    // ---- 18 per-genre micro-MLPs, scalar (200 MACs/lane)
    if (lane < NG) {
        const int g = lane;
        float gin0 = normg[g], gin1 = 1.f, gin2 = 0.f, gin3 = 1.f - normg[g];
        float a1[16];
        #pragma unroll
        for (int o = 0; o < 16; ++o) {
            const float* w = Wa1 + g * 64 + o * 4;
            float s = ba1[g * 16 + o] + w[0] * gin0 + w[1] * gin1 + w[2] * gin2 + w[3] * gin3;
            a1[o] = fmaxf(s, 0.f);
        }
        float a2[8];
        #pragma unroll
        for (int o = 0; o < 8; ++o) {
            float s = ba2[g * 8 + o];
            const float* w = Wa2 + g * 128 + o * 16;
            #pragma unroll
            for (int i = 0; i < 16; ++i) s += w[i] * a1[i];
            a2[o] = fmaxf(s, 0.f);
        }
        float s = ba3[g];
        #pragma unroll
        for (int i = 0; i < 8; ++i) s += Wa3[g * 8 + i] * a2[i];
        float a3 = 1.f / (1.f + expf(-s));

        float deficit = 1.f / (float)NG - normg[g];
        float mult = (deficit > 0.f) ? (1.f + deficit) : (1.f + 0.5f * deficit);
        float ga = fminf(fmaxf(a3 * mult, 0.1f), 2.f);
        out[g] = fminf(fmaxf(mainAdj[g] * ga, 0.1f), 2.f);
    }
    if (lane == 0) out[NG] = itemGiniSh;
}

// ---------------------------------------------------------------- launch
extern "C" void kernel_launch(void* const* d_in, const int* in_sizes, int n_in,
                              void* d_out, int out_size, void* d_ws, size_t ws_size,
                              hipStream_t stream) {
    const float* gcounts = (const float*)d_in[0];
    const float* items   = (const float*)d_in[1];
    const float* W1f = (const float*)d_in[2];
    const float* b1f = (const float*)d_in[3];
    const float* lng = (const float*)d_in[4];
    const float* lnb = (const float*)d_in[5];
    const float* W2f = (const float*)d_in[6];
    const float* b2f = (const float*)d_in[7];
    const float* W3f = (const float*)d_in[8];
    const float* b3f = (const float*)d_in[9];
    const float* Wa1 = (const float*)d_in[10];
    const float* ba1 = (const float*)d_in[11];
    const float* Wa2 = (const float*)d_in[12];
    const float* ba2 = (const float*)d_in[13];
    const float* Wa3 = (const float*)d_in[14];
    const float* ba3 = (const float*)d_in[15];
    float* out = (float*)d_out;

    char* ws = (char*)d_ws;
    unsigned* cnt = (unsigned*)ws;                                   // 8 MB
    double*   chT = (double*)(ws + (size_t)NBINS * 4);               // 8 KB
    double*   chS = (double*)(ws + (size_t)NBINS * 4 + 8192);        // 8 KB
    unsigned* chC = (unsigned*)(ws + (size_t)NBINS * 4 + 16384);     // 4 KB
    float*    acc = (float*)(ws + (size_t)NBINS * 4 + 16384 + 4096); // 8 B

    k_zero<<<2048, 256, 0, stream>>>(cnt, acc);
    k_hist<<<4096, 256, 0, stream>>>(items, cnt, acc);
    k_chunkscan<<<NCHUNK, 256, 0, stream>>>(cnt, chT, chS, chC);
    k_final<<<1, 32, 0, stream>>>(gcounts, chT, chS, chC, acc,
                                  W1f, b1f, lng, lnb, W2f, b2f, W3f, b3f,
                                  Wa1, ba1, Wa2, ba2, Wa3, ba3, out);
}
